// DirectionalMamba2D_7559142441333
// MI455X (gfx1250) — matrix operand order repaired and checked by reading, not compiled, not run on hardware
//
#include <hip/hip_runtime.h>
#include <math.h>

typedef __attribute__((ext_vector_type(16))) __bf16 v16bf;
typedef __attribute__((ext_vector_type(8)))  float  v8f;
typedef __attribute__((ext_vector_type(4)))  unsigned int v4u;
typedef __attribute__((ext_vector_type(8)))  unsigned int v8u;

#define NTOK   16777216ull          // B*H*W*C = 8*128*128*128 elements (token-major)
#define TOKENS 131072ull            // B*H*W tokens
#define C      128
#define C3     384

// ---- WMMA fragment helpers (contiguous k order used by this file, same for A and B) ----
// A (16x32 bf16): lane l: M=l&15, ksel=l>>4; elems 0..15 -> K=ksel*16+e (16 contiguous)
__device__ __forceinline__ v16bf load_a16(const __bf16* p) {
  v16bf a;
#pragma unroll
  for (int e = 0; e < 8; ++e) { a[e] = p[e]; a[e + 8] = p[e + 8]; }
  return a;
}
// B (32x16 bf16, row-major [N][K] staging): lane l: N=l&15, K = (l>>4)*16 + e (16 contiguous)
__device__ __forceinline__ v16bf load_b16(const __bf16* p) {
  return *(const v16bf*)p;  // 32B aligned LDS load
}
__device__ __forceinline__ v8f wmma_bf16(v16bf a, v16bf b, v8f c) {
  return __builtin_amdgcn_wmma_f32_16x16x32_bf16(false, a, false, b, (short)0, c, false, false);
}
__device__ __forceinline__ float wred(float v) {
#pragma unroll
  for (int off = 16; off > 0; off >>= 1) v += __shfl_xor(v, off, 32);
  return v;
}
__device__ __forceinline__ float sigm(float x) { return 1.0f / (1.0f + __expf(-x)); }

// ---- Tensor Data Mover: async 2D tile (16 rows x 768B) global -> LDS ----
// D# packing per CDNA5 ISA: group0 {count=1, lds_addr,
// global_addr, type=2}; group1 {data_size=4B, tensor_dim0=192, tensor_dim1=16,
// tile_dim0=192, tile_dim1=16, tensor_dim0_stride=strideUnits}. Groups 2/3
// zero (dims >=2 unused). Tracked with TENSORcnt.
__device__ __forceinline__ void tdm_load_tile(unsigned lds_off, unsigned long long gaddr,
                                              unsigned strideUnits) {
  v4u g0; v8u g1; v4u gz = {0u, 0u, 0u, 0u};
  g0[0] = 1u;                                   // count=1, user descriptor
  g0[1] = lds_off;                              // lds_addr (bytes)
  g0[2] = (unsigned)gaddr;                      // global_addr[31:0]
  g0[3] = ((unsigned)(gaddr >> 32) & 0x01FFFFFFu) | (2u << 30);  // addr[56:32] | type=2
  g1[0] = (2u << 16);                           // workgroup_mask=0, data_size=2 (4B units)
  g1[1] = (192u << 16);                         // tensor_dim0[15:0]=192 (in bits 63:48)
  g1[2] = (16u << 16);                          // tensor_dim0 hi=0 | tensor_dim1 lo=16
  g1[3] = (192u << 16);                         // tensor_dim1 hi=0 | tile_dim0=192
  g1[4] = 16u;                                  // tile_dim1=16, tile_dim2=0
  g1[5] = strideUnits;                          // tensor_dim0_stride[31:0] (4B units)
  g1[6] = 0u;                                   // stride hi | dim1_stride lo
  g1[7] = 0u;
  asm volatile("tensor_load_to_lds %0, %1, %2, %3"
               :: "s"(g0), "s"(g1), "s"(gz), "s"(gz) : "memory");
}

// ============ Kernel 1: x [B,C,H,W] -> xT [B,H,W,C] (fp32 + bf16) ============
__global__ void k_transpose(const float* __restrict__ x, float* __restrict__ xTf,
                            __bf16* __restrict__ xTb) {
  __shared__ float tile[32][33];
  const int bh = blockIdx.x;             // b*128+h
  const int cblk = blockIdx.y, wblk = blockIdx.z;
  const int tx = threadIdx.x & 31, ty = threadIdx.x >> 5;
  const int b = bh >> 7, hh = bh & 127;
  const size_t planeHW = 128 * 128;
  const size_t base = (size_t)b * 128 * planeHW + (size_t)hh * 128;
#pragma unroll
  for (int i = 0; i < 4; ++i) {
    const int cl = ty + 8 * i;
    tile[cl][tx] = x[base + (size_t)(cblk * 32 + cl) * planeHW + (wblk * 32 + tx)];
  }
  __syncthreads();
#pragma unroll
  for (int i = 0; i < 4; ++i) {
    const int wl = ty + 8 * i;
    const int w = wblk * 32 + wl;
    const int c = cblk * 32 + tx;
    const size_t idx = ((size_t)bh * 128 + w) * 128 + c;
    const float v = tile[tx][wl];
    xTf[idx] = v;
    xTb[idx] = (__bf16)v;
  }
}

// ============ Kernel 2: xw = xT * Wih^T + bih  ([TOKENS,128] x [128,384]) ============
__global__ void k_xw_gemm(const __bf16* __restrict__ xTb, const float* __restrict__ Wih,
                          const float* __restrict__ bih, __bf16* __restrict__ xw) {
  __shared__ __align__(32) __bf16 sW[C3 * C];   // 98304 B, Wih row-major [384][128]
  __shared__ __align__(32) __bf16 sA[16 * C];   // 4096 B, 16-token tile
  const int tid = threadIdx.x;
  for (int i = tid; i < C3 * C; i += 256) sW[i] = (__bf16)Wih[i];
  const size_t tok0 = (size_t)blockIdx.x * 16;
  for (int i = tid; i < 16 * C; i += 256) sA[i] = xTb[tok0 * C + i];
  __syncthreads();
  const int wv = tid >> 5, ln = tid & 31;
  const int nl = ln & 15, ksel = ln >> 4;
#pragma unroll
  for (int nt = 0; nt < 3; ++nt) {          // 8 waves * 3 tiles = 24 N-tiles = 384
    const int n16 = (wv * 3 + nt) * 16;
    v8f acc = {};
#pragma unroll
    for (int kb = 0; kb < 4; ++kb) {
      v16bf a = load_a16(&sA[nl * C + kb * 32 + ksel * 16]);
      v16bf bmat = load_b16(&sW[(n16 + nl) * C + kb * 32 + ksel * 16]);
      acc = wmma_bf16(a, bmat, acc);
    }
    const int o = n16 + nl;
    const float bb = bih[o];
#pragma unroll
    for (int v = 0; v < 8; ++v) {
      const int m = v + ksel * 8;
      xw[(tok0 + m) * C3 + o] = (__bf16)(acc[v] + bb);
    }
  }
}

// ============ Kernel 3: GRU scan (4 directions, 16 sequences per block) ============
// Per step: TDM prefetches next step's 16x384 bf16 xw tile into an LDS double
// buffer (overlapped with this step's 96 WMMAs), then fused gating from LDS.
__global__ void k_scan(const __bf16* __restrict__ xw_h, const __bf16* __restrict__ xw_v,
                       const float* __restrict__ Whh_h, const float* __restrict__ bhh_h,
                       const float* __restrict__ Whh_v, const float* __restrict__ bhh_v,
                       float* __restrict__ gru) {
  __shared__ __align__(32) __bf16 sW[C3 * C];      // 98304 B  Whh (bf16) [384][128]
  __shared__ float  sH[16 * C];                    // 8192  B  h (fp32)
  __shared__ __align__(32) __bf16 sHb[16 * C];     // 4096  B  h (bf16, A operand)
  __shared__ float  sHW[16 * C3];                  // 24576 B  hw = h*Whh^T + bhh
  __shared__ float  sBhh[C3];                      // 1536  B
  __shared__ __align__(16) __bf16 sXW[2][16 * C3]; // 24576 B  TDM double buffer
  const int dir = blockIdx.y;
  const float* Whh = (dir < 2) ? Whh_h : Whh_v;
  const float* bhh = (dir < 2) ? bhh_h : bhh_v;
  const __bf16* xw = (dir < 2) ? xw_h : xw_v;
  float* out = gru + (size_t)dir * NTOK;
  const int tid = threadIdx.x;
  for (int i = tid; i < C3 * C; i += 256) sW[i] = (__bf16)Whh[i];
  for (int i = tid; i < C3; i += 256) sBhh[i] = bhh[i];
  for (int i = tid; i < 16 * C; i += 256) { sH[i] = 0.0f; sHb[i] = (__bf16)0.0f; }
  const int wv = tid >> 5, ln = tid & 31;
  const int nl = ln & 15, ksel = ln >> 4;
  const int seqBase = blockIdx.x * 16;
  const int b0 = seqBase >> 7, q0 = seqBase & 127;   // q = row (dirs 0/1) or col (dirs 2/3)

  auto tdm_issue = [&](int tt) {
    size_t token0;
    if (dir == 0)      token0 = (size_t)b0 * 16384 + (size_t)q0 * 128 + tt;
    else if (dir == 1) token0 = (size_t)b0 * 16384 + (size_t)q0 * 128 + (127 - tt);
    else if (dir == 2) token0 = (size_t)b0 * 16384 + (size_t)tt * 128 + q0;
    else               token0 = (size_t)b0 * 16384 + (size_t)(127 - tt) * 128 + q0;
    const unsigned long long ga = (unsigned long long)(uintptr_t)xw + token0 * 768ull;
    const unsigned lds_off = (unsigned)(uintptr_t)&sXW[tt & 1][0];
    tdm_load_tile(lds_off, ga, (dir < 2) ? 24576u : 192u);  // row stride in dwords
  };

  if (wv == 0) tdm_issue(0);   // prime the pipeline (TDM ignores EXEC; wave-uniform)
  __syncthreads();

  for (int t = 0; t < 128; ++t) {
    if (wv == 0 && t + 1 < 128) tdm_issue(t + 1);   // prefetch next step
    // --- hw = h @ Whh^T + bhh : 24 N-tiles over 8 waves, K=128 ---
#pragma unroll
    for (int nt = 0; nt < 3; ++nt) {
      const int n16 = (wv * 3 + nt) * 16;
      v8f acc = {};
#pragma unroll
      for (int kb = 0; kb < 4; ++kb) {
        v16bf a = load_a16(&sHb[nl * C + kb * 32 + ksel * 16]);
        v16bf bmat = load_b16(&sW[(n16 + nl) * C + kb * 32 + ksel * 16]);
        acc = wmma_bf16(a, bmat, acc);
      }
      const int o = n16 + nl;
      const float bb = sBhh[o];
#pragma unroll
      for (int v = 0; v < 8; ++v) sHW[(v + ksel * 8) * C3 + o] = acc[v] + bb;
    }
    // wave 0 waits for this step's tile (<=1 outstanding keeps t+1 in flight)
    if (wv == 0) {
      if (t + 1 < 128) __builtin_amdgcn_s_wait_tensorcnt(1);
      else             __builtin_amdgcn_s_wait_tensorcnt(0);
    }
    __syncthreads();   // publishes TDM-written LDS + sHW to all waves
    // --- fused GRU gating; each thread owns 8 (seq,chan) elements ---
    const __bf16* xbuf = &sXW[t & 1][0];
#pragma unroll
    for (int i = 0; i < 8; ++i) {
      const int e = tid + 256 * i;
      const int s = e >> 7, c = e & 127;
      const float xr = (float)xbuf[s * C3 + c];
      const float xz = (float)xbuf[s * C3 + c + 128];
      const float xn = (float)xbuf[s * C3 + c + 256];
      const float hr = sHW[s * C3 + c], hz = sHW[s * C3 + c + 128], hn = sHW[s * C3 + c + 256];
      const float hprev = sH[s * C + c];
      const float r = sigm(xr + hr);
      const float z = sigm(xz + hz);
      const float n = tanhf(xn + r * hn);
      const float hN = (1.0f - z) * n + z * hprev;
      size_t token;
      {
        const int seq = seqBase + s;
        const int b = seq >> 7, q = seq & 127;
        if (dir == 0)      token = (size_t)b * 16384 + (size_t)q * 128 + t;
        else if (dir == 1) token = (size_t)b * 16384 + (size_t)q * 128 + (127 - t);
        else if (dir == 2) token = (size_t)b * 16384 + (size_t)t * 128 + q;
        else               token = (size_t)b * 16384 + (size_t)(127 - t) * 128 + q;
      }
      out[token * C + c] = hN;
      sH[s * C + c] = hN;
      sHb[s * C + c] = (__bf16)hN;
    }
    __syncthreads();
  }
}

// ============ Kernel 4: LN x4 -> folded 1x1 conv (WMMA) -> BN -> GELU -> +x -> LN ============
__global__ void k_fuse(const float* __restrict__ gru, const float* __restrict__ xTf,
                       const float* __restrict__ lng_h, const float* __restrict__ lnb_h,
                       const float* __restrict__ lng_v, const float* __restrict__ lnb_v,
                       const float* __restrict__ conv_w, const float* __restrict__ conv_b,
                       const float* __restrict__ bn_g, const float* __restrict__ bn_b,
                       const float* __restrict__ bn_m, const float* __restrict__ bn_v,
                       const float* __restrict__ ln_g, const float* __restrict__ ln_b,
                       float* __restrict__ outp) {
  __shared__ __align__(32) __bf16 sWe[C * 256]; // 65536 B, folded weight [n=out][k=in]
  __shared__ __align__(32) __bf16 sF[16 * 256]; // 8192  B, features [out_h | out_v]
  __shared__ float sP[16 * C];                  // 8192  B, post-GELU activations
  const int tid = threadIdx.x;
  // cat is [h,v,h,v] -> fold conv_w [128,512] to [128,256]
  for (int i = tid; i < C * 256; i += 256) {
    const int n = i >> 8, k = i & 255;
    const float w = (k < 128)
        ? conv_w[n * 512 + k] + conv_w[n * 512 + 256 + k]
        : conv_w[n * 512 + 128 + (k - 128)] + conv_w[n * 512 + 384 + (k - 128)];
    sWe[i] = (__bf16)w;
  }
  const size_t tok0 = (size_t)blockIdx.x * 16;
  const int wv = tid >> 5, ln = tid & 31;
  // --- per-direction LayerNorm + 0.5*(fwd+bwd) merge; wave handles 2 tokens ---
#pragma unroll
  for (int tt = 0; tt < 2; ++tt) {
    const int m = wv * 2 + tt;
    const size_t token = tok0 + m;
    const float* xc = xTf + token * C;
    float v0[4], v1[4], v2[4], v3[4];
    float s0 = 0, s1 = 0, s2 = 0, s3 = 0, q0 = 0, q1 = 0, q2 = 0, q3 = 0;
#pragma unroll
    for (int j = 0; j < 4; ++j) {
      const int c = ln * 4 + j;
      const float xv = xc[c];
      v0[j] = gru[0 * NTOK + token * C + c] + xv;
      v1[j] = gru[1 * NTOK + token * C + c] + xv;
      v2[j] = gru[2 * NTOK + token * C + c] + xv;
      v3[j] = gru[3 * NTOK + token * C + c] + xv;
      s0 += v0[j]; q0 += v0[j] * v0[j];
      s1 += v1[j]; q1 += v1[j] * v1[j];
      s2 += v2[j]; q2 += v2[j] * v2[j];
      s3 += v3[j]; q3 += v3[j] * v3[j];
    }
    s0 = wred(s0); q0 = wred(q0); s1 = wred(s1); q1 = wred(q1);
    s2 = wred(s2); q2 = wred(q2); s3 = wred(s3); q3 = wred(q3);
    const float k1 = 1.0f / 128.0f;
    const float mu0 = s0 * k1, mu1 = s1 * k1, mu2 = s2 * k1, mu3 = s3 * k1;
    const float r0 = rsqrtf(q0 * k1 - mu0 * mu0 + 1e-5f);
    const float r1 = rsqrtf(q1 * k1 - mu1 * mu1 + 1e-5f);
    const float r2 = rsqrtf(q2 * k1 - mu2 * mu2 + 1e-5f);
    const float r3 = rsqrtf(q3 * k1 - mu3 * mu3 + 1e-5f);
#pragma unroll
    for (int j = 0; j < 4; ++j) {
      const int c = ln * 4 + j;
      const float gh = lng_h[c], bh = lnb_h[c], gv = lng_v[c], bv = lnb_v[c];
      const float oh = 0.5f * (((v0[j] - mu0) * r0 * gh + bh) + ((v1[j] - mu1) * r1 * gh + bh));
      const float ov = 0.5f * (((v2[j] - mu2) * r2 * gv + bv) + ((v3[j] - mu3) * r3 * gv + bv));
      sF[m * 256 + c] = (__bf16)oh;
      sF[m * 256 + 128 + c] = (__bf16)ov;
    }
  }
  __syncthreads();
  // --- conv: [16,256] x [256,128]; wave wv owns one 16-wide N-tile ---
  const int nl = ln & 15, ksel = ln >> 4;
  v8f acc = {};
#pragma unroll
  for (int kb = 0; kb < 8; ++kb) {
    v16bf a = load_a16(&sF[nl * 256 + kb * 32 + ksel * 16]);
    v16bf bmat = load_b16(&sWe[(wv * 16 + nl) * 256 + kb * 32 + ksel * 16]);
    acc = wmma_bf16(a, bmat, acc);
  }
  {
    const int o = wv * 16 + nl;
    const float rs = rsqrtf(bn_v[o] + 1e-5f);
    const float g = bn_g[o], bb = bn_b[o], mn = bn_m[o], cb = conv_b[o];
#pragma unroll
    for (int v = 0; v < 8; ++v) {
      const int m = v + ksel * 8;
      float val = acc[v] + cb;
      val = (val - mn) * rs * g + bb;
      val = 0.5f * val * (1.0f + erff(val * 0.70710678118654752f));  // exact GELU
      val += xTf[(tok0 + m) * C + o];
      sP[m * C + o] = val;
    }
  }
  __syncthreads();
  // --- final LayerNorm over channels + transposed store to [B,C,H,W] ---
#pragma unroll
  for (int tt = 0; tt < 2; ++tt) {
    const int m = wv * 2 + tt;
    const size_t token = tok0 + m;
    float vv[4], s = 0, q = 0;
#pragma unroll
    for (int j = 0; j < 4; ++j) {
      vv[j] = sP[m * C + ln * 4 + j];
      s += vv[j]; q += vv[j] * vv[j];
    }
    s = wred(s); q = wred(q);
    const float mu = s * (1.0f / 128.0f);
    const float rv = rsqrtf(q * (1.0f / 128.0f) - mu * mu + 1e-5f);
    const int b = (int)(token >> 14), hh = (int)((token >> 7) & 127), ww = (int)(token & 127);
#pragma unroll
    for (int j = 0; j < 4; ++j) {
      const int c = ln * 4 + j;
      const float y = (vv[j] - mu) * rv * ln_g[c] + ln_b[c];
      outp[(((size_t)b * 128 + c) * 128 + hh) * 128 + ww] = y;
    }
  }
}

extern "C" void kernel_launch(void* const* d_in, const int* in_sizes, int n_in,
                              void* d_out, int out_size, void* d_ws, size_t ws_size,
                              hipStream_t stream) {
  (void)in_sizes; (void)n_in; (void)out_size; (void)ws_size;
  const float* x      = (const float*)d_in[0];
  const float* Wih_h  = (const float*)d_in[1];
  const float* Whh_h  = (const float*)d_in[2];
  const float* bih_h  = (const float*)d_in[3];
  const float* bhh_h  = (const float*)d_in[4];
  const float* lng_h  = (const float*)d_in[5];
  const float* lnb_h  = (const float*)d_in[6];
  const float* Wih_v  = (const float*)d_in[7];
  const float* Whh_v  = (const float*)d_in[8];
  const float* bih_v  = (const float*)d_in[9];
  const float* bhh_v  = (const float*)d_in[10];
  const float* lng_v  = (const float*)d_in[11];
  const float* lnb_v  = (const float*)d_in[12];
  const float* conv_w = (const float*)d_in[13];
  const float* conv_b = (const float*)d_in[14];
  const float* bn_g   = (const float*)d_in[15];
  const float* bn_b   = (const float*)d_in[16];
  const float* bn_m   = (const float*)d_in[17];
  const float* bn_v   = (const float*)d_in[18];
  const float* ln_g   = (const float*)d_in[19];
  const float* ln_b   = (const float*)d_in[20];
  float* outp = (float*)d_out;

  // Workspace carve-up (~544 MiB total)
  char* p = (char*)d_ws;
  float*  xTf  = (float*)p;  p += NTOK * 4;           // [B,H,W,C] fp32
  __bf16* xTb  = (__bf16*)p; p += NTOK * 2;           // [B,H,W,C] bf16
  __bf16* xwh  = (__bf16*)p; p += NTOK * 3 * 2;       // [tok,384] bf16
  __bf16* xwv  = (__bf16*)p; p += NTOK * 3 * 2;       // [tok,384] bf16
  float*  gruo = (float*)p;  p += 4 * NTOK * 4;       // 4 x [tok,128] fp32

  k_transpose<<<dim3(1024, 4, 4), 256, 0, stream>>>(x, xTf, xTb);
  k_xw_gemm<<<dim3(TOKENS / 16), 256, 0, stream>>>(xTb, Wih_h, bih_h, xwh);
  k_xw_gemm<<<dim3(TOKENS / 16), 256, 0, stream>>>(xTb, Wih_v, bih_v, xwv);
  k_scan<<<dim3(64, 4), 256, 0, stream>>>(xwh, xwv, Whh_h, bhh_h, Whh_v, bhh_v, gruo);
  k_fuse<<<dim3(TOKENS / 16), 256, 0, stream>>>(gruo, xTf, lng_h, lnb_h, lng_v, lnb_v,
                                                conv_w, conv_b, bn_g, bn_b, bn_m, bn_v,
                                                ln_g, ln_b, outp);
}
